// PatchCoreWrapper_77129022702057
// MI455X (gfx1250) — matrix operand order repaired and checked by reading, not compiled, not run on hardware
//
#include <hip/hip_runtime.h>
#include <hip/hip_bf16.h>
#include <math.h>

typedef __attribute__((ext_vector_type(16))) __bf16 v16bf;
typedef __attribute__((ext_vector_type(8)))  __bf16 v8bf;
typedef __attribute__((ext_vector_type(4)))  __bf16 v4bf;
typedef __attribute__((ext_vector_type(8)))  float  v8f;
typedef __attribute__((ext_vector_type(4)))  float  f4;

#define BI  16      // batch
#define HH  48
#define WW  48
#define CC  1536
#define C2  512
#define C3  1024
#define MM  16384   // memory rows
#define NQ  36864   // total queries (16*48*48)

#define QT  32                       // queries per workgroup (2 N-tiles)
#define LDSQ_BYTES  (QT * CC * 2)    // 98304
#define LDST_BYTES  (256 * 2 * 9 * 4)// 18432
#define KNN_LDS     (LDSQ_BYTES + LDST_BYTES)

// ---------------------------------------------------------------- init
__global__ void k_init(float* __restrict__ img) {
    if (threadIdx.x < BI) img[threadIdx.x] = 0.0f;
}

// ------------------------------------------------- memory bank -> bf16
__global__ void k_cvt_mb(const float* __restrict__ mb, __bf16* __restrict__ out, int n4) {
    int i = blockIdx.x * blockDim.x + threadIdx.x;
    if (i >= n4) return;
    f4 v = ((const f4*)mb)[i];
    v4bf o;
    o.x = (__bf16)v.x; o.y = (__bf16)v.y; o.z = (__bf16)v.z; o.w = (__bf16)v.w;
    ((v4bf*)out)[i] = o;
}

// ------------------- upsample feat3 + concat + L2 normalize -> bf16 Q
// one block = 8 consecutive pixels (same row), 256 threads
__global__ void k_embed(const float* __restrict__ f2, const float* __restrict__ f3,
                        __bf16* __restrict__ qmat) {
    __shared__ float emb[8 * CC];     // 48KB
    __shared__ float psum[8][32];
    __shared__ float invn[8];

    int t = threadIdx.x;
    int p = t & 7, q = t >> 3;        // pixel-in-tile, channel group
    int pix0 = blockIdx.x * 8;
    int b   = pix0 / (HH * WW);
    int rem = pix0 % (HH * WW);
    int y   = rem / WW;
    int x   = (rem % WW) + p;

    // half-pixel bilinear 48 -> 24 source coords (align_corners=False)
    float sx = x * 0.5f - 0.25f, sy = y * 0.5f - 0.25f;
    int ix = (int)floorf(sx), iy = (int)floorf(sy);
    float fx = sx - (float)ix, fy = sy - (float)iy;
    int x0c = min(max(ix, 0), 23),     x1c = min(max(ix + 1, 0), 23);
    int y0c = min(max(iy, 0), 23),     y1c = min(max(iy + 1, 0), 23);

    float ss = 0.0f;
    for (int cb = 0; cb < 48; ++cb) {
        int c = q * 48 + cb;
        float v;
        if (c < C2) {
            v = f2[(((size_t)b * C2 + c) * HH + y) * WW + x];
        } else {
            int c3 = c - C2;
            const float* base = f3 + ((size_t)b * C3 + c3) * 24 * 24;
            float v00 = base[y0c * 24 + x0c], v01 = base[y0c * 24 + x1c];
            float v10 = base[y1c * 24 + x0c], v11 = base[y1c * 24 + x1c];
            v = (1.0f - fy) * ((1.0f - fx) * v00 + fx * v01)
              +         fy  * ((1.0f - fx) * v10 + fx * v11);
        }
        emb[p * CC + c] = v;
        ss += v * v;
    }
    psum[p][q] = ss;
    __syncthreads();
    if (t < 8) {
        float s = 0.0f;
        for (int j = 0; j < 32; ++j) s += psum[t][j];
        invn[t] = 1.0f / fmaxf(sqrtf(s), 1e-12f);
    }
    __syncthreads();
    // write normalized bf16, coalesced over channels
    for (int i = 0; i < 48; ++i) {
        int pp = i / 6;
        int c  = (i % 6) * 256 + t;
        qmat[(size_t)(pix0 + pp) * CC + c] = (__bf16)(emb[pp * CC + c] * invn[pp]);
    }
}

// ----------------------------------------------------- top-9 insertion
__device__ __forceinline__ void topk_insert(float (&tp)[9], float v) {
#pragma unroll
    for (int i = 0; i < 9; ++i) {
        float hi = fmaxf(tp[i], v);
        v        = fminf(tp[i], v);
        tp[i]    = hi;
    }
}

// -------------------------- fused GEMM (bf16 WMMA) + per-query top-9
// block = 256 threads (8 waves) owns 32 queries (two WMMA N-tiles -> each
// streamed A fragment feeds 2 WMMAs, halving L2 traffic per FLOP).
// Each wave handles 4 M-tiles (64 memory rows) per step; 32 steps cover 16384.
__global__ void __launch_bounds__(256)
k_knn(const __bf16* __restrict__ qmat, const __bf16* __restrict__ mbb,
      float* __restrict__ out_patch, float* __restrict__ out_img) {
    extern __shared__ __align__(32) char smem[];
    __bf16* lds_q  = (__bf16*)smem;                  // 96KB: 32 query rows
    float*  lds_top = (float*)(smem + LDSQ_BYTES);   // 18KB: 256 lanes * 2 lists * 9

    int t = threadIdx.x;
    int qbase = blockIdx.x * QT;

    // stage 32 query rows into LDS (8B per thread per iter, coalesced)
    {
        const uint2* src = (const uint2*)(qmat + (size_t)qbase * CC);
        uint2* dst = (uint2*)lds_q;
#pragma unroll
        for (int i = 0; i < 48; ++i) dst[i * 256 + t] = src[i * 256 + t];
    }
    __syncthreads();

    int wave = t >> 5, lane = t & 31;
    int half = lane >> 4, ln = lane & 15;

    float tp0[9], tp1[9];
#pragma unroll
    for (int i = 0; i < 9; ++i) { tp0[i] = -3.0e38f; tp1[i] = -3.0e38f; }

    // B fragment sources: query columns ln and 16+ln, K chunk = k0 + 16*half
    const __bf16* qrow0 = lds_q + ln * CC + half * 16;
    const __bf16* qrow1 = lds_q + (16 + ln) * CC + half * 16;

    for (int step = 0; step < 32; ++step) {
        int mbase = step * 512 + wave * 64;
        v8f zero = {};
        v8f acc0[4], acc1[4];
        const __bf16* arow[4];
#pragma unroll
        for (int u = 0; u < 4; ++u) {
            acc0[u] = zero;
            acc1[u] = zero;
            // A fragment source: memory row = mbase + u*16 + ln, K chunk = k0 + 16*half
            arow[u] = mbb + (size_t)(mbase + u * 16 + ln) * CC + half * 16;
            // prefetch next step's rows toward this WGP (global_prefetch_b8)
            if (step < 31) __builtin_prefetch(arow[u] + 512 * CC, 0, 1);
        }
        for (int ks = 0; ks < 48; ++ks) {
            int k0 = ks * 32;
            v16bf b0 = *(const v16bf*)(qrow0 + k0);   // 32B contiguous LDS reads
            v16bf b1 = *(const v16bf*)(qrow1 + k0);
#pragma unroll
            for (int u = 0; u < 4; ++u) {
                union { v16bf v; v8bf h[2]; } a;
                a.h[0] = *(const v8bf*)(arow[u] + k0);
                a.h[1] = *(const v8bf*)(arow[u] + k0 + 8);
                acc0[u] = __builtin_amdgcn_wmma_f32_16x16x32_bf16(
                    false, a.v, false, b0, (short)0, acc0[u], false, false);
                acc1[u] = __builtin_amdgcn_wmma_f32_16x16x32_bf16(
                    false, a.v, false, b1, (short)0, acc1[u], false, false);
            }
        }
        // lane's candidates: acc0 -> query (qbase+ln), acc1 -> query (qbase+16+ln)
#pragma unroll
        for (int u = 0; u < 4; ++u) {
#pragma unroll
            for (int r = 0; r < 8; ++r) {
                topk_insert(tp0, acc0[u][r]);
                topk_insert(tp1, acc1[u][r]);
            }
        }
    }

#pragma unroll
    for (int i = 0; i < 9; ++i) {
        lds_top[(t * 2 + 0) * 9 + i] = tp0[i];
        lds_top[(t * 2 + 1) * 9 + i] = tp1[i];
    }
    __syncthreads();

    if (t < QT) {
        int nt = t >> 4, ln2 = t & 15;
        float fin[9];
#pragma unroll
        for (int i = 0; i < 9; ++i) fin[i] = -3.0e38f;
        // merge lists from both wave-halves of all 8 waves with matching lane
        for (int s = 0; s < 16; ++s) {
            int src = (s >> 1) * 32 + (s & 1) * 16 + ln2;
            const float* lst = &lds_top[(src * 2 + nt) * 9];
            for (int i = 0; i < 9; ++i) topk_insert(fin, lst[i]);
        }
        float accs = 0.0f;
#pragma unroll
        for (int i = 0; i < 9; ++i) {
            float d2 = fmaxf(2.0f - 2.0f * fin[i], 0.0f);
            accs += sqrtf(d2);
        }
        float sc = accs * (1.0f / 9.0f);
        int qi = qbase + t;
        out_patch[qi] = sc;
        int bimg = qi / (HH * WW);
        // scores are non-negative -> uint compare == float compare
        atomicMax((unsigned int*)(out_img + bimg), __float_as_uint(sc));
    }
}

// --------------------------------------- separable 25-tap gaussian blur
__device__ __forceinline__ int reflect48(int i) {
    i = (i < 0) ? -i : i;
    return (i > 47) ? (94 - i) : i;
}

__global__ void k_blur_h(const float* __restrict__ in, float* __restrict__ out) {
    int idx = blockIdx.x * blockDim.x + threadIdx.x;
    if (idx >= NQ) return;
    int x = idx % WW, rest = idx / WW;   // rest = b*48 + y
    float w[25], s = 0.0f;
#pragma unroll
    for (int i = 0; i < 25; ++i) { float d = (float)(i - 12); w[i] = expf(-d * d * (1.0f / 32.0f)); s += w[i]; }
    float inv = 1.0f / s, acc = 0.0f;
#pragma unroll
    for (int i = 0; i < 25; ++i) acc += w[i] * in[rest * WW + reflect48(x - 12 + i)];
    out[idx] = acc * inv;
}

__global__ void k_blur_v(const float* __restrict__ in, float* __restrict__ out) {
    int idx = blockIdx.x * blockDim.x + threadIdx.x;
    if (idx >= NQ) return;
    int x = idx % WW, rest = idx / WW;
    int y = rest % HH, b = rest / HH;
    float w[25], s = 0.0f;
#pragma unroll
    for (int i = 0; i < 25; ++i) { float d = (float)(i - 12); w[i] = expf(-d * d * (1.0f / 32.0f)); s += w[i]; }
    float inv = 1.0f / s, acc = 0.0f;
#pragma unroll
    for (int i = 0; i < 25; ++i) acc += w[i] * in[(b * HH + reflect48(y - 12 + i)) * WW + x];
    out[idx] = acc * inv;
}

// ---------------------------------------------------------------- host
extern "C" void kernel_launch(void* const* d_in, const int* in_sizes, int n_in,
                              void* d_out, int out_size, void* d_ws, size_t ws_size,
                              hipStream_t stream) {
    const float* f2 = (const float*)d_in[0];   // [16,512,48,48]
    const float* f3 = (const float*)d_in[1];   // [16,1024,24,24]
    const float* mb = (const float*)d_in[2];   // [16384,1536]

    float* out       = (float*)d_out;
    float* out_img   = out;                 // [16]
    float* out_amap  = out + BI;            // [16*48*48]
    float* out_patch = out + BI + NQ;       // [36864]

    char* ws = (char*)d_ws;
    const size_t Q_BYTES  = (size_t)NQ * CC * 2;   // 113,246,208
    const size_t MB_BYTES = (size_t)MM * CC * 2;   //  50,331,648
    __bf16* qmat = (__bf16*)ws;
    __bf16* mbb  = (__bf16*)(ws + Q_BYTES);
    float*  tmp  = (float*)(ws + Q_BYTES + MB_BYTES);

    // allow 114KB of dynamic LDS for the fused GEMM kernel (CDNA5: 320KB/WGP)
    static bool attr_set = false;
    hipFuncSetAttribute((const void*)k_knn,
                        hipFuncAttributeMaxDynamicSharedMemorySize, KNN_LDS);

    k_init  <<<1, 32, 0, stream>>>(out_img);
    {
        int n4 = (MM * CC) / 4;   // 6,291,456
        k_cvt_mb<<<(n4 + 255) / 256, 256, 0, stream>>>(mb, mbb, n4);
    }
    k_embed <<<NQ / 8, 256, 0, stream>>>(f2, f3, qmat);
    k_knn   <<<NQ / QT, 256, KNN_LDS, stream>>>(qmat, mbb, out_patch, out_img);
    k_blur_h<<<(NQ + 255) / 256, 256, 0, stream>>>(out_patch, tmp);
    k_blur_v<<<(NQ + 255) / 256, 256, 0, stream>>>(tmp, out_amap);
}
